// NodeModel_25598005084722
// MI455X (gfx1250) — matrix operand order repaired and checked by reading, not compiled, not run on hardware
//
#include <hip/hip_runtime.h>

// ---------------------------------------------------------------------------
// GraphNet NodeModel for MI455X (gfx1250), wave32 + WMMA f16 (16x16x32).
//
// Sizes (compile-time constants from the reference):
//   N=10000 nodes, E=320000 edges, F_N=F_E=128, F_G=32, LATENT=32, F_OUT=128
//
// All GEMMs run on v_wmma_f32_16x16x32_f16. One wave owns one 16-row tile.
// Weights are pre-swizzled once per call into per-lane B-fragments in d_ws.
// Hidden activations (16x32 f16) ping-pong through LDS to convert the WMMA
// D-layout (col-per-lane) into the A-layout (row-per-lane).
// ---------------------------------------------------------------------------

typedef __attribute__((ext_vector_type(16))) _Float16 v16h;
typedef __attribute__((ext_vector_type(8)))  float    v8f;
typedef __attribute__((ext_vector_type(8)))  unsigned v8u;

#define NNODES 10000
#define NEDGES 320000
#define FN 128
#define FE 128
#define FG 32
#define FOUT 128

// fragment-table bases (each fragment = 32 lanes x 8 dwords = 1KB)
#define F_M1W0 0    // 256x32  -> 8 kt * 2 nt = 16 frags
#define F_M1W1 16   // 32x32   -> 2
#define F_M1W2 18   // 32x32   -> 2
#define F_M1W3 20   // 32x256  -> 16
#define F_M2W0 36   // 416x32  -> 13 * 2 = 26
#define F_M2W1 62   // 32x32   -> 2
#define F_M2W2 64   // 32x32   -> 2
#define F_M2W3 66   // 32x128  -> 8
#define F_TOTAL 74

union AFragU {
  v16h h;
  uint4 q[2];
  v8u  u;
};

__device__ __forceinline__ v8f wmma16x16x32(v16h a, v16h b, v8f c) {
  // 8 args: (neg_a, A, neg_b, B, c_mod, C, reuse_a, reuse_b)
  return __builtin_amdgcn_wmma_f32_16x16x32_f16(false, a, false, b,
                                                (short)0, c, false, false);
}

// B fragment: pre-swizzled in workspace; 2x global_load_b128 per operand.
__device__ __forceinline__ v16h load_bfrag(const unsigned* __restrict__ wf,
                                           int frag, int lane) {
  AFragU u;
  u.u = ((const v8u*)wf)[(size_t)frag * 32 + lane];
  return u.h;
}

// A fragment (16x32, f16) from fp32 global memory, with optional scale.
// ISA layout: lane holds row m=lane&15; element j -> K = (j<8 ? j : j+8)+8*half
// => two runs of 8 consecutive floats at K = 8*half and K = 16+8*half.
__device__ __forceinline__ v16h load_a_f32(const float* __restrict__ src,
                                           int half, float scale) {
  const float4* p = (const float4*)src;
  float4 f0 = p[2 * half + 0];
  float4 f1 = p[2 * half + 1];
  float4 f2 = p[2 * half + 4];
  float4 f3 = p[2 * half + 5];
  v16h a;
  a[0]  = (_Float16)(f0.x * scale); a[1]  = (_Float16)(f0.y * scale);
  a[2]  = (_Float16)(f0.z * scale); a[3]  = (_Float16)(f0.w * scale);
  a[4]  = (_Float16)(f1.x * scale); a[5]  = (_Float16)(f1.y * scale);
  a[6]  = (_Float16)(f1.z * scale); a[7]  = (_Float16)(f1.w * scale);
  a[8]  = (_Float16)(f2.x * scale); a[9]  = (_Float16)(f2.y * scale);
  a[10] = (_Float16)(f2.z * scale); a[11] = (_Float16)(f2.w * scale);
  a[12] = (_Float16)(f3.x * scale); a[13] = (_Float16)(f3.y * scale);
  a[14] = (_Float16)(f3.z * scale); a[15] = (_Float16)(f3.w * scale);
  return a;
}

// A fragment from an LDS 16x32 f16 tile (row-major): two ds_load_b128.
__device__ __forceinline__ v16h load_a_lds(const _Float16* hb, int lane) {
  int m = lane & 15, half = lane >> 4;
  const uint4* p = (const uint4*)(hb + m * 32);
  AFragU u;
  u.q[0] = p[half];      // K =  8*half .. +7
  u.q[1] = p[2 + half];  // K = 16+8*half .. +7
  return u.h;
}

// Apply bias + ReLU to the two 16x16 accumulators and store the 16x32 f16
// tile into LDS. D layout: lane holds col n=lane&15, VGPR r -> row r+8*half.
__device__ __forceinline__ void store_h_relu(_Float16* hb, v8f c0, v8f c1,
                                             const float* __restrict__ bias,
                                             int lane) {
  int n = lane & 15, half = lane >> 4;
  float b0 = bias[n], b1 = bias[16 + n];
#pragma unroll
  for (int r = 0; r < 8; ++r) {
    int mr = r + 8 * half;
    hb[mr * 32 + n]      = (_Float16)fmaxf(c0[r] + b0, 0.0f);
    hb[mr * 32 + 16 + n] = (_Float16)fmaxf(c1[r] + b1, 0.0f);
  }
}

// ---------------------------------------------------------------------------
__global__ void zero_f32(float* __restrict__ p, long n) {
  long i = (long)blockIdx.x * blockDim.x + threadIdx.x;
  long stride = (long)gridDim.x * blockDim.x;
  for (; i < n; i += stride) p[i] = 0.0f;
}

// Pre-swizzle one (kt,nt) 32x16 f16 B-fragment per 32-thread block.
// W is row-major [Kdim][Ndim]. B layout: N = lane&15, K(j) = (j<8 ? j : j+8) + 8*(lane>>4).
__global__ void wprep(const float* __restrict__ W, int Ndim, int nTilesN,
                      unsigned* __restrict__ wf, int fragBase) {
  int lane = threadIdx.x;
  int kt = blockIdx.x / nTilesN;
  int nt = blockIdx.x % nTilesN;
  int n = lane & 15, half = lane >> 4;
  v16h b;
#pragma unroll
  for (int j = 0; j < 16; ++j) {
    int K = kt * 32 + (j < 8 ? j : j + 8) + 8 * half;
    b[j] = (_Float16)W[(size_t)K * Ndim + nt * 16 + n];
  }
  AFragU u;
  u.h = b;
  ((v8u*)wf)[(size_t)(fragBase + blockIdx.x) * 32 + lane] = u.u;
}

// ---------------------------------------------------------------------------
// Edge path: h = MLP1(concat(x[row], edge_attr)); scatter-add into s[col],
// count into cnt[col]. 8 waves/block, one 16-edge tile per wave.
__global__ __launch_bounds__(256) void edge_kernel(
    const float* __restrict__ x, const int* __restrict__ ei,
    const float* __restrict__ ea, const unsigned* __restrict__ wf,
    const float* __restrict__ b1, const float* __restrict__ b2,
    const float* __restrict__ b3, const float* __restrict__ b4,
    float* __restrict__ s, float* __restrict__ cnt) {
  __shared__ __attribute__((aligned(16))) _Float16 hbuf[2][8][16 * 32];
  __shared__ int colbuf[8][16];

  int tid = threadIdx.x, wave = tid >> 5, lane = tid & 31;
  int tile = blockIdx.x * 8 + wave;
  int e0 = tile * 16;
  if (e0 >= NEDGES) return;
  int m = lane & 15, half = lane >> 4;

  int row_m = ei[e0 + m];           // sender node of edge (e0+m)
  int col_m = ei[NEDGES + e0 + m];  // receiver node
  if (half == 0) colbuf[wave][m] = col_m;

  // ---- layer 1: 256 -> 32 (A built straight from global fp32) ----
  v8f c0 = {}, c1 = {};
#pragma unroll
  for (int kt = 0; kt < 8; ++kt) {
    const float* src = (kt < 4)
        ? x + (size_t)row_m * FN + kt * 32
        : ea + (size_t)(e0 + m) * FE + (kt - 4) * 32;
    v16h a = load_a_f32(src, half, 1.0f);
    c0 = wmma16x16x32(a, load_bfrag(wf, F_M1W0 + kt * 2 + 0, lane), c0);
    c1 = wmma16x16x32(a, load_bfrag(wf, F_M1W0 + kt * 2 + 1, lane), c1);
  }
  store_h_relu(hbuf[0][wave], c0, c1, b1, lane);
  __syncthreads();

  // ---- layer 2: 32 -> 32 ----
  {
    v16h a = load_a_lds(hbuf[0][wave], lane);
    v8f z = {};
    c0 = wmma16x16x32(a, load_bfrag(wf, F_M1W1 + 0, lane), z);
    c1 = wmma16x16x32(a, load_bfrag(wf, F_M1W1 + 1, lane), z);
  }
  store_h_relu(hbuf[1][wave], c0, c1, b2, lane);
  __syncthreads();

  // ---- layer 3: 32 -> 32 ----
  {
    v16h a = load_a_lds(hbuf[1][wave], lane);
    v8f z = {};
    c0 = wmma16x16x32(a, load_bfrag(wf, F_M1W2 + 0, lane), z);
    c1 = wmma16x16x32(a, load_bfrag(wf, F_M1W2 + 1, lane), z);
  }
  store_h_relu(hbuf[0][wave], c0, c1, b3, lane);
  __syncthreads();

  // ---- layer 4: 32 -> 256, ReLU, fused scatter-add ----
  v16h a3 = load_a_lds(hbuf[0][wave], lane);
  if (half == 0) atomicAdd(&cnt[col_m], 1.0f);
  int n = m;
#pragma unroll
  for (int nt = 0; nt < 16; ++nt) {
    v8f z = {};
    v8f d = wmma16x16x32(a3, load_bfrag(wf, F_M1W3 + nt, lane), z);
    float bn = b4[nt * 16 + n];
#pragma unroll
    for (int r = 0; r < 8; ++r) {
      int mr = r + 8 * half;
      float v = fmaxf(d[r] + bn, 0.0f);
      atomicAdd(&s[(size_t)colbuf[wave][mr] * 256 + nt * 16 + n], v);
    }
  }
}

// ---------------------------------------------------------------------------
// Node path: out = MLP2(concat(x, s/max(cnt,1), g)), no final activation.
__global__ __launch_bounds__(256) void node_kernel(
    const float* __restrict__ x, const float* __restrict__ s,
    const float* __restrict__ cnt, const float* __restrict__ g,
    const unsigned* __restrict__ wf,
    const float* __restrict__ b1, const float* __restrict__ b2,
    const float* __restrict__ b3, const float* __restrict__ b4,
    float* __restrict__ out) {
  __shared__ __attribute__((aligned(16))) _Float16 hbuf[2][8][16 * 32];

  int tid = threadIdx.x, wave = tid >> 5, lane = tid & 31;
  int tile = blockIdx.x * 8 + wave;
  int i0 = tile * 16;
  if (i0 >= NNODES) return;
  int m = lane & 15, half = lane >> 4;

  float inv = 1.0f / fmaxf(cnt[i0 + m], 1.0f);  // scatter-mean divisor

  // ---- layer 1: 416 -> 32 (13 k-tiles: x | s*inv | g broadcast) ----
  v8f c0 = {}, c1 = {};
#pragma unroll
  for (int kt = 0; kt < 13; ++kt) {
    const float* src;
    float scale = 1.0f;
    if (kt < 4) {
      src = x + (size_t)(i0 + m) * FN + kt * 32;
    } else if (kt < 12) {
      src = s + (size_t)(i0 + m) * 256 + (kt - 4) * 32;
      scale = inv;
    } else {
      src = g;  // same 32 global feats for every row
    }
    v16h a = load_a_f32(src, half, scale);
    c0 = wmma16x16x32(a, load_bfrag(wf, F_M2W0 + kt * 2 + 0, lane), c0);
    c1 = wmma16x16x32(a, load_bfrag(wf, F_M2W0 + kt * 2 + 1, lane), c1);
  }
  store_h_relu(hbuf[0][wave], c0, c1, b1, lane);
  __syncthreads();

  // ---- layer 2 ----
  {
    v16h a = load_a_lds(hbuf[0][wave], lane);
    v8f z = {};
    c0 = wmma16x16x32(a, load_bfrag(wf, F_M2W1 + 0, lane), z);
    c1 = wmma16x16x32(a, load_bfrag(wf, F_M2W1 + 1, lane), z);
  }
  store_h_relu(hbuf[1][wave], c0, c1, b2, lane);
  __syncthreads();

  // ---- layer 3 ----
  {
    v16h a = load_a_lds(hbuf[1][wave], lane);
    v8f z = {};
    c0 = wmma16x16x32(a, load_bfrag(wf, F_M2W2 + 0, lane), z);
    c1 = wmma16x16x32(a, load_bfrag(wf, F_M2W2 + 1, lane), z);
  }
  store_h_relu(hbuf[0][wave], c0, c1, b3, lane);
  __syncthreads();

  // ---- layer 4: 32 -> 128, bias only, store ----
  v16h a3 = load_a_lds(hbuf[0][wave], lane);
  int n = m;
#pragma unroll
  for (int nt = 0; nt < 8; ++nt) {
    v8f z = {};
    v8f d = wmma16x16x32(a3, load_bfrag(wf, F_M2W3 + nt, lane), z);
    float bn = b4[nt * 16 + n];
#pragma unroll
    for (int r = 0; r < 8; ++r) {
      int mr = r + 8 * half;
      out[(size_t)(i0 + mr) * FOUT + nt * 16 + n] = d[r] + bn;
    }
  }
}

// ---------------------------------------------------------------------------
extern "C" void kernel_launch(void* const* d_in, const int* in_sizes, int n_in,
                              void* d_out, int out_size, void* d_ws,
                              size_t ws_size, hipStream_t stream) {
  (void)in_sizes; (void)n_in; (void)out_size; (void)ws_size;

  const float* x    = (const float*)d_in[0];
  const int*   ei   = (const int*)d_in[1];   // [2,E] row-major: row then col
  const float* ea   = (const float*)d_in[2];
  const float* g    = (const float*)d_in[3];
  const float* m1w0 = (const float*)d_in[4];
  const float* m1b0 = (const float*)d_in[5];
  const float* m1w1 = (const float*)d_in[6];
  const float* m1b1 = (const float*)d_in[7];
  const float* m1w2 = (const float*)d_in[8];
  const float* m1b2 = (const float*)d_in[9];
  const float* m1w3 = (const float*)d_in[10];
  const float* m1b3 = (const float*)d_in[11];
  const float* m2w0 = (const float*)d_in[12];
  const float* m2b0 = (const float*)d_in[13];
  const float* m2w1 = (const float*)d_in[14];
  const float* m2b1 = (const float*)d_in[15];
  const float* m2w2 = (const float*)d_in[16];
  const float* m2b2 = (const float*)d_in[17];
  const float* m2w3 = (const float*)d_in[18];
  const float* m2b3 = (const float*)d_in[19];
  float* out = (float*)d_out;

  // workspace layout: s[N*256] f32 | cnt[N] f32 | wfrag (74KB, 32B aligned)
  float* s   = (float*)d_ws;
  float* cnt = s + (size_t)NNODES * 256;
  unsigned* wf = (unsigned*)(cnt + NNODES);  // byte off 10,280,000 (32B mult)

  zero_f32<<<1024, 256, 0, stream>>>(s, (long)NNODES * 256 + NNODES);

  // Pre-swizzle all weight matrices into WMMA B-fragments.
  wprep<<<16, 32, 0, stream>>>(m1w0,  32,  2, wf, F_M1W0);
  wprep<<<2,  32, 0, stream>>>(m1w1,  32,  2, wf, F_M1W1);
  wprep<<<2,  32, 0, stream>>>(m1w2,  32,  2, wf, F_M1W2);
  wprep<<<16, 32, 0, stream>>>(m1w3, 256, 16, wf, F_M1W3);
  wprep<<<26, 32, 0, stream>>>(m2w0,  32,  2, wf, F_M2W0);
  wprep<<<2,  32, 0, stream>>>(m2w1,  32,  2, wf, F_M2W1);
  wprep<<<2,  32, 0, stream>>>(m2w2,  32,  2, wf, F_M2W2);
  wprep<<<8,  32, 0, stream>>>(m2w3, 128,  8, wf, F_M2W3);

  // 20000 edge tiles / 8 waves = 2500 blocks; 625 node tiles -> 79 blocks.
  edge_kernel<<<(NEDGES / 16 + 7) / 8, 256, 0, stream>>>(
      x, ei, ea, wf, m1b0, m1b1, m1b2, m1b3, s, cnt);
  node_kernel<<<((NNODES + 15) / 16 + 7) / 8, 256, 0, stream>>>(
      x, s, cnt, g, wf, m2b0, m2b1, m2b2, m2b3, out);
}
